// attention_9663676416592
// MI455X (gfx1250) — matrix operand order repaired and checked by reading, not compiled, not run on hardware
//
#include <hip/hip_runtime.h>
#include <hip/hip_bf16.h>

// ---------------------------------------------------------------------------
// MI455X (gfx1250) fused correlation-attention.
//
// Reference reduces to 4 attention instances (2 batches x 2 directions) with
// head_dim=256 (bf16 WMMA contraction) and value_dim=1 (scalar v[j]=Wg.X[:,j]).
// Dominant cost: 34.4 GFLOP of logits GEMM -> v_wmma_f32_16x16x32_bf16.
// All tensors L2-resident; flash-style online softmax avoids ever spilling the
// 2x(4096x4096) logit matrices (would be 268MB) to memory.
// Key-tile staging uses gfx1250 async global->LDS loads (ASYNCcnt) when the
// toolchain exposes them, falling back to global->VGPR->ds_store otherwise.
// ---------------------------------------------------------------------------

#define AD   4096   // T*H*W
#define CDIM 256    // channels

typedef __attribute__((ext_vector_type(16))) __bf16 v16bf;
typedef __attribute__((ext_vector_type(8)))  float  v8f;

#if __has_builtin(__builtin_amdgcn_global_load_async_to_lds_b128)
#define USE_ASYNC_LDS 1
// builtin expects native clang vector element type (not HIP_vector_type)
typedef __attribute__((vector_size(16))) int nv4i;
typedef __attribute__((address_space(1))) nv4i as1_v4i;
typedef __attribute__((address_space(3))) nv4i as3_v4i;
#else
#define USE_ASYNC_LDS 0
#endif

__device__ __forceinline__ void wait_async0() {
#if __has_builtin(__builtin_amdgcn_s_wait_asynccnt)
  __builtin_amdgcn_s_wait_asynccnt(0);
#else
  asm volatile("s_wait_asynccnt 0x0" ::: "memory");
#endif
}

union FragU { uint4 u[2]; v16bf v; };

__device__ __forceinline__ v16bf make_frag(uint4 a, uint4 b) {
  FragU f; f.u[0] = a; f.u[1] = b; return f.v;
}

// ---------------------------------------------------------------------------
// Pass 1: transpose+convert  X[b,c,j] (f32) -> XT[b,j,c] (bf16), and
//         v[b,j] = sum_c Wg[c] * X[b,c,j].
// grid: 2 (m) * 2 (b) * 64 (j-tiles of 64) = 256 blocks, 256 threads.
// ---------------------------------------------------------------------------
__global__ __launch_bounds__(256) void prep_kernel(
    const float* __restrict__ ex, const float* __restrict__ qu,
    const float* __restrict__ Wg,
    __bf16* __restrict__ XT_e, __bf16* __restrict__ XT_q,
    float* __restrict__ v_ex, float* __restrict__ v_q)
{
  __shared__ float Xs[CDIM * 66];            // 256 rows, pad 64->66 (67.6KB)
  const int blk = blockIdx.x;
  const int jt  = blk & 63;
  const int bm  = blk >> 6;                  // 0..3
  const int b   = bm & 1, m = bm >> 1;
  const float*  X    = (m == 0 ? ex : qu) + (size_t)b * CDIM * AD;
  __bf16*       XT   = (m == 0 ? XT_e : XT_q) + (size_t)b * AD * CDIM;
  float*        vout = (m == 0 ? v_ex : v_q) + b * AD;
  const int t = threadIdx.x;

  // load 256(c) x 64(j) f32 tile, coalesced float4 along j
  #pragma unroll 4
  for (int it = 0; it < 16; ++it) {
    int cid = t + 256 * it;                  // 0..4095 float4 chunks
    int c = cid >> 4, f = cid & 15;
    float4 d = *(const float4*)(X + (size_t)c * AD + jt * 64 + f * 4);
    float* dst = &Xs[c * 66 + f * 4];
    dst[0] = d.x; dst[1] = d.y; dst[2] = d.z; dst[3] = d.w;
  }
  __syncthreads();

  // transposed write: row j is 256 contiguous bf16 (512B, coalesced)
  for (int j = 0; j < 64; ++j)
    XT[(size_t)(jt * 64 + j) * CDIM + t] = (__bf16)Xs[t * 66 + j];

  // v[j] = Wg . X[:,j]
  if (t < 64) {
    float acc = 0.f;
    for (int c = 0; c < CDIM; ++c) acc += Wg[c] * Xs[c * 66 + t];
    vout[jt * 64 + t] = acc;
  }
}

// ---------------------------------------------------------------------------
// Pass 1b: W (f32 [o][k]) -> bf16. 2 x 65536 elements.
// ---------------------------------------------------------------------------
__global__ __launch_bounds__(256) void wconv_kernel(
    const float* __restrict__ We, const float* __restrict__ Wq,
    __bf16* __restrict__ WbfE, __bf16* __restrict__ WbfQ)
{
  int i = blockIdx.x * 256 + threadIdx.x;
  WbfE[i] = (__bf16)We[i];
  WbfQ[i] = (__bf16)Wq[i];
}

// ---------------------------------------------------------------------------
// Pass 2: corr[b,i,o] = sum_k XT[b,i,k] * W[o,k]  (bf16 in, f32 acc, bf16 out)
// grid: 2*2*(AD/128) = 128 blocks, 8 waves, wave = one 16-row M tile x 16 N tiles.
// ---------------------------------------------------------------------------
__global__ __launch_bounds__(256) void proj_kernel(
    const __bf16* __restrict__ XT_e, const __bf16* __restrict__ XT_q,
    const __bf16* __restrict__ Wbf_e, const __bf16* __restrict__ Wbf_q,
    __bf16* __restrict__ corr_e, __bf16* __restrict__ corr_q)
{
  const int blk = blockIdx.x;
  const int rowblk = blk & 31;               // AD/128
  const int bm = blk >> 5;
  const int b = bm & 1, m = bm >> 1;
  const __bf16* XT  = (m == 0 ? XT_e : XT_q) + (size_t)b * AD * CDIM;
  const __bf16* W   = (m == 0 ? Wbf_e : Wbf_q);
  __bf16*       cor = (m == 0 ? corr_e : corr_q) + (size_t)b * AD * CDIM;

  const int t = threadIdx.x, wave = t >> 5, lane = t & 31;
  const int lj = lane & 15, kh = lane >> 4;
  const int i0 = rowblk * 128 + wave * 16;

  // preload A fragments for this wave's 16 rows (8 k-steps of 32)
  uint4 afr0[8], afr1[8];
  {
    const char* qrow = (const char*)(XT + (size_t)(i0 + lj) * CDIM);
    #pragma unroll
    for (int ks = 0; ks < 8; ++ks) {
      const uint4* p = (const uint4*)(qrow + ks * 64 + kh * 16);
      afr0[ks] = p[0];
      afr1[ks] = p[2];                       // +32 bytes (K+16 chunk)
    }
  }

  for (int nt = 0; nt < 16; ++nt) {
    v8f acc;
    #pragma unroll
    for (int e = 0; e < 8; ++e) acc[e] = 0.f;
    const char* wrow = (const char*)(W + (size_t)(nt * 16 + lj) * CDIM);
    #pragma unroll
    for (int ks = 0; ks < 8; ++ks) {
      const uint4* bp = (const uint4*)(wrow + ks * 64 + kh * 16);
      v16bf bv = make_frag(bp[0], bp[2]);
      v16bf av = make_frag(afr0[ks], afr1[ks]);
      acc = __builtin_amdgcn_wmma_f32_16x16x32_bf16(
          false, av, false, bv, (short)0, acc, false, false);
    }
    // D layout: VGPR v -> row v (lanes 0-15) / v+8 (lanes 16-31), col = lane%16
    #pragma unroll
    for (int v = 0; v < 8; ++v)
      cor[(size_t)(i0 + v + 8 * kh) * CDIM + nt * 16 + lj] = (__bf16)acc[v];
  }
}

// ---------------------------------------------------------------------------
// Pass 3: flash attention, value_dim = 1.
//   out[i] = sigmoid( sum_j softmax_j(S[i,j]) * v[j] ),  S = corr . KT^T
// grid: 4 instances * 32 row-blocks = 128 blocks, 256 thr (8 waves x 16 rows).
// Key tiles (64 j x 256 c bf16) double-buffered in LDS via async global->LDS
// loads; rows padded to 264 bf16 (33 uint4) for conflict-free ds_load_b128.
// ---------------------------------------------------------------------------
__global__ __launch_bounds__(256) void flash_kernel(
    const __bf16* __restrict__ corr_e, const __bf16* __restrict__ corr_q,
    const __bf16* __restrict__ XT_e,  const __bf16* __restrict__ XT_q,
    const float* __restrict__ v_ex,   const float* __restrict__ v_q,
    float* __restrict__ out)
{
  __shared__ uint4 lds[2][64 * 33];          // 2 x 33.8KB
  const int inst   = blockIdx.x >> 5;        // 0..3
  const int rowblk = blockIdx.x & 31;
  const int m = inst >> 1, b = inst & 1;
  const __bf16* Q  = (m == 0 ? corr_e : corr_q) + (size_t)b * AD * CDIM;
  const __bf16* KT = (m == 0 ? XT_q   : XT_e)   + (size_t)b * AD * CDIM;
  const float*  vv = (m == 0 ? v_ex   : v_q)    + b * AD;
  float*        o  = out + (size_t)m * 2 * AD + (size_t)b * AD;

  const int t = threadIdx.x, wave = t >> 5, lane = t & 31;
  const int lj = lane & 15, kh = lane >> 4;
  const int i0 = rowblk * 128 + wave * 16;
  const float LOG2E = 1.44269504089f;

  // preload this wave's A fragments (16 query rows x 256 K, bf16)
  uint4 afr0[8], afr1[8];
  {
    const char* qrow = (const char*)(Q + (size_t)(i0 + lj) * CDIM);
    #pragma unroll
    for (int ks = 0; ks < 8; ++ks) {
      const uint4* p = (const uint4*)(qrow + ks * 64 + kh * 16);
      afr0[ks] = p[0];
      afr1[ks] = p[2];
    }
  }

  float runM[8], runL[8], runO[8];
  #pragma unroll
  for (int v = 0; v < 8; ++v) { runM[v] = -1e30f; runL[v] = 0.f; runO[v] = 0.f; }

#if USE_ASYNC_LDS
  // each thread DMAs 8 x 16B chunks of the 64x256 bf16 key tile straight
  // into LDS; completion tracked by ASYNCcnt (no staging VGPRs, no ds_store)
  auto stage_async = [&](int jt, int bufi) {
    const uint4* src = (const uint4*)(KT + (size_t)jt * 64 * CDIM);
    #pragma unroll
    for (int it = 0; it < 8; ++it) {
      int cid = t + 256 * it;                // 64 rows x 32 chunks
      as1_v4i* gp = (as1_v4i*)&src[(cid >> 5) * 32 + (cid & 31)];
      as3_v4i* lp = (as3_v4i*)&lds[bufi][(cid >> 5) * 33 + (cid & 31)];
      __builtin_amdgcn_global_load_async_to_lds_b128(gp, lp, 0, 0);
    }
  };

  stage_async(0, 0);
  wait_async0();
  __syncthreads();
#else
  uint4 greg[8];
  auto stage_load = [&](int jt) {
    const uint4* src = (const uint4*)(KT + (size_t)jt * 64 * CDIM);
    #pragma unroll
    for (int it = 0; it < 8; ++it) {
      int cid = t + 256 * it;
      greg[it] = src[(cid >> 5) * 32 + (cid & 31)];
    }
  };
  auto stage_store = [&](int bufi) {
    #pragma unroll
    for (int it = 0; it < 8; ++it) {
      int cid = t + 256 * it;
      lds[bufi][(cid >> 5) * 33 + (cid & 31)] = greg[it];
    }
  };
  stage_load(0);
  stage_store(0);
  __syncthreads();
#endif

  for (int jt = 0; jt < 64; ++jt) {
#if USE_ASYNC_LDS
    // safe: the barrier ending jt-1 proved all waves finished reading this buf
    if (jt + 1 < 64) stage_async(jt + 1, (jt + 1) & 1);
#else
    if (jt + 1 < 64) stage_load(jt + 1);
#endif

    // ---- S = Q-rows x 64 key columns (4 N tiles, 8 K steps) ----
    v8f acc[4];
    #pragma unroll
    for (int nt = 0; nt < 4; ++nt)
      #pragma unroll
      for (int e = 0; e < 8; ++e) acc[nt][e] = 0.f;

    const uint4* buf = &lds[jt & 1][0];
    #pragma unroll
    for (int ks = 0; ks < 8; ++ks) {
      v16bf av = make_frag(afr0[ks], afr1[ks]);
      #pragma unroll
      for (int nt = 0; nt < 4; ++nt) {
        const uint4* bp = buf + (nt * 16 + lj) * 33 + ks * 4 + kh;
        v16bf bv = make_frag(bp[0], bp[2]);
        acc[nt] = __builtin_amdgcn_wmma_f32_16x16x32_bf16(
            false, av, false, bv, (short)0, acc[nt], false, false);
      }
    }

    // ---- online softmax + value accumulation ----
    float cv0 = vv[jt * 64 +  0 + lj];
    float cv1 = vv[jt * 64 + 16 + lj];
    float cv2 = vv[jt * 64 + 32 + lj];
    float cv3 = vv[jt * 64 + 48 + lj];

    #pragma unroll
    for (int v = 0; v < 8; ++v) {
      float s0 = acc[0][v], s1 = acc[1][v], s2 = acc[2][v], s3 = acc[3][v];
      float mx = fmaxf(fmaxf(s0, s1), fmaxf(s2, s3));
      #pragma unroll
      for (int off = 1; off < 16; off <<= 1)
        mx = fmaxf(mx, __shfl_xor(mx, off, 32));     // row lives in 16 lanes
      float newM = fmaxf(runM[v], mx);
      float p0 = exp2f((s0 - newM) * LOG2E);
      float p1 = exp2f((s1 - newM) * LOG2E);
      float p2 = exp2f((s2 - newM) * LOG2E);
      float p3 = exp2f((s3 - newM) * LOG2E);
      float ps = p0 + p1 + p2 + p3;
      float pw = p0 * cv0 + p1 * cv1 + p2 * cv2 + p3 * cv3;
      #pragma unroll
      for (int off = 1; off < 16; off <<= 1) {
        ps += __shfl_xor(ps, off, 32);
        pw += __shfl_xor(pw, off, 32);
      }
      float cf = exp2f((runM[v] - newM) * LOG2E);
      runL[v] = runL[v] * cf + ps;
      runO[v] = runO[v] * cf + pw;
      runM[v] = newM;
    }

#if USE_ASYNC_LDS
    wait_async0();                            // my async writes for jt+1 done
    __syncthreads();                          // everyone's writes visible
#else
    __syncthreads();
    if (jt + 1 < 64) {
      stage_store((jt + 1) & 1);
      __syncthreads();
    }
#endif
  }

  // row value replicated across each 16-lane half; lane 0/16 writes
  if ((lane & 15) == 0) {
    #pragma unroll
    for (int v = 0; v < 8; ++v) {
      float val = runO[v] / runL[v];
      float sig = 1.f / (1.f + exp2f(-val * LOG2E));
      o[i0 + v + 8 * kh] = sig;
    }
  }
}

// ---------------------------------------------------------------------------
extern "C" void kernel_launch(void* const* d_in, const int* in_sizes, int n_in,
                              void* d_out, int out_size, void* d_ws, size_t ws_size,
                              hipStream_t stream) {
  const float* ex = (const float*)d_in[0];   // [2,256,4,32,32]
  const float* qu = (const float*)d_in[1];
  const float* We = (const float*)d_in[2];   // [256,256]
  const float* Wq = (const float*)d_in[3];
  const float* Wg = (const float*)d_in[4];   // [256]
  float* out = (float*)d_out;                // m1 [2,4096] ++ m2 [2,4096]

  const size_t SZ = (size_t)2 * AD * CDIM;   // elements per [B,AD,C] tensor
  __bf16* XT_e   = (__bf16*)d_ws;            // bf16 transposed flats (= keys)
  __bf16* XT_q   = XT_e + SZ;
  __bf16* corr_e = XT_q + SZ;                // bf16 projected queries
  __bf16* corr_q = corr_e + SZ;
  __bf16* Wbf_e  = corr_q + SZ;
  __bf16* Wbf_q  = Wbf_e + (size_t)CDIM * CDIM;
  float*  v_ex   = (float*)(Wbf_q + (size_t)CDIM * CDIM);
  float*  v_q    = v_ex + 2 * AD;            // total ~16.6 MB of workspace

  prep_kernel <<<256, 256, 0, stream>>>(ex, qu, Wg, XT_e, XT_q, v_ex, v_q);
  wconv_kernel<<<256, 256, 0, stream>>>(We, Wq, Wbf_e, Wbf_q);
  proj_kernel <<<128, 256, 0, stream>>>(XT_e, XT_q, Wbf_e, Wbf_q, corr_e, corr_q);
  flash_kernel<<<128, 256, 0, stream>>>(corr_e, corr_q, XT_e, XT_q, v_ex, v_q, out);
}
